// MultiHeadAttention_45664092291425
// MI455X (gfx1250) — matrix operand order repaired and checked by reading, not compiled, not run on hardware
//
#include <hip/hip_runtime.h>
#include <hip/hip_bf16.h>

// MHA forward for B=2,S=2048,C=1024,H=16,D=64 on gfx1250 (wave32).
// GEMMs via v_wmma_f32_16x16x32_bf16 (f32 accumulate).
// d_out = [ out f32 (2*2048*1024) | attn f32 (2*16*2048*2048) ].
// d_ws: 4 bf16 buffers of 4M elems = 32 MB total.
// Fused attention: scores+softmax+attn@V in one kernel; the 537 MB attn
// tensor is streamed LDS->HBM with global_store_async_from_lds_b128 so the
// store drains while the attn@V WMMA loop runs.

typedef __bf16 bf16_t;
typedef bf16_t v16bf __attribute__((ext_vector_type(16)));
typedef float  v8f   __attribute__((ext_vector_type(8)));

#define C_IN   1024
#define SEQ    2048
#define NH     16
#define DH     64
#define SC_LD  2052            // 2048 + 4 pad: kills LDS bank conflicts

__device__ inline v8f wmma_bf16(v16bf a, v16bf b, v8f c) {
  // 8 args: (neg_a, A, neg_b, B, c_mod, C, reuse_a, reuse_b)
  return __builtin_amdgcn_wmma_f32_16x16x32_bf16(false, a, false, b, (short)0, c,
                                                 false, false);
}

// Async LDS -> global copy, 16 bytes per lane, tracked by ASYNCcnt.
__device__ inline void async_store_b128(const void* gptr, unsigned lds_off) {
  asm volatile("global_store_async_from_lds_b128 %0, %1, off"
               :: "v"(gptr), "v"(lds_off)
               : "memory");
}

// ---- fragment loaders -----------------------------------------------------
// A (16x32 bf16): lane m = lane&15; lanes 0-15 hold K {kb..kb+7, kb+16..kb+23}
// with kb=0, lanes 16-31 the same with kb=8.
__device__ inline v16bf load_a_bf16(const bf16_t* __restrict__ base, int m0,
                                    int ldk, int k0, int lane) {
  const int mr = lane & 15;
  const int kb = k0 + ((lane & 16) ? 8 : 0);
  const bf16_t* p = base + (size_t)(m0 + mr) * ldk + kb;
  union { v16bf v; uint4 q[2]; } f;
  f.q[0] = *(const uint4*)(p);        // K = kb .. kb+7
  f.q[1] = *(const uint4*)(p + 16);   // K = kb+16 .. kb+23
  return f.v;
}

// B (32x16 bf16): lane col n = lane&15; lanes 0-15 hold K {kb..kb+7, kb+16..kb+23}
// with kb=0, lanes 16-31 the same with kb=8.  Source: "row n holds K contiguously".
__device__ inline v16bf load_b_bf16(const bf16_t* __restrict__ base, int n0,
                                    int ldk, int k0, int lane) {
  const int nr = lane & 15;
  const int kb = k0 + ((lane & 16) ? 8 : 0);
  const bf16_t* p = base + (size_t)(n0 + nr) * ldk + kb;
  union { v16bf v; uint4 q[2]; } f;
  f.q[0] = *(const uint4*)(p);
  f.q[1] = *(const uint4*)(p + 16);
  return f.v;
}

__device__ inline void cvt8(v16bf& d, int off, const float* __restrict__ p) {
  float4 a = ((const float4*)p)[0];
  float4 b = ((const float4*)p)[1];
  d[off + 0] = (bf16_t)a.x; d[off + 1] = (bf16_t)a.y;
  d[off + 2] = (bf16_t)a.z; d[off + 3] = (bf16_t)a.w;
  d[off + 4] = (bf16_t)b.x; d[off + 5] = (bf16_t)b.y;
  d[off + 6] = (bf16_t)b.z; d[off + 7] = (bf16_t)b.w;
}

__device__ inline v16bf load_a_f32(const float* __restrict__ base, int m0,
                                   int ldk, int k0, int lane) {
  const int mr = lane & 15;
  const int kb = k0 + ((lane & 16) ? 8 : 0);
  const float* p = base + (size_t)(m0 + mr) * ldk + kb;
  v16bf d;
  cvt8(d, 0, p);
  cvt8(d, 8, p + 16);
  return d;
}

__device__ inline v16bf load_b_f32(const float* __restrict__ base, int n0,
                                   int ldk, int k0, int lane) {
  const int nr = lane & 15;
  const int kb = k0 + ((lane & 16) ? 8 : 0);
  const float* p = base + (size_t)(n0 + nr) * ldk + kb;
  v16bf d;
  cvt8(d, 0, p);
  cvt8(d, 8, p + 16);
  return d;
}

// A-fragment from f32 values resident in LDS (softmax rows), cvt to bf16.
__device__ inline v16bf lds_load_a_f32(const float* sc, int ld, int k0,
                                       int lane) {
  const int mr = lane & 15;
  const int kb = k0 + ((lane & 16) ? 8 : 0);
  const float* p = sc + mr * ld + kb;
  v16bf d;
#pragma unroll
  for (int i = 0; i < 8; ++i) d[i] = (bf16_t)p[i];
#pragma unroll
  for (int i = 0; i < 8; ++i) d[8 + i] = (bf16_t)p[16 + i];
  return d;
}

// ---- kernel 1: fused QKV projection + RoPE --------------------------------
// grid = (32, 16 heads, 3 {q,k,v}); block = 256 (8 waves).
__global__ __launch_bounds__(256) void qkv_rope_kernel(
    const float* __restrict__ q, const float* __restrict__ k,
    const float* __restrict__ v, const float* __restrict__ Wq,
    const float* __restrict__ Wk, const float* __restrict__ Wv,
    bf16_t* __restrict__ qh, bf16_t* __restrict__ kh,
    bf16_t* __restrict__ vhT) {
  const int lane  = threadIdx.x & 31;
  const int wave  = threadIdx.x >> 5;
  const int which = blockIdx.z;                       // 0=q 1=k 2=v
  const int m0    = (blockIdx.x * 8 + wave) * 16;     // row tile (m = b*S+s)
  const int h     = blockIdx.y;                       // head; n0 = h*64

  const float* X = (which == 0) ? q : (which == 1) ? k : v;
  const float* W = (which == 0) ? Wq : (which == 1) ? Wk : Wv;

  v8f acc[4] = {};
  for (int k0 = 0; k0 < C_IN; k0 += 32) {
    v16bf a = load_a_f32(X, m0, C_IN, k0, lane);
#pragma unroll
    for (int t = 0; t < 4; ++t) {
      v16bf b = load_b_f32(W, h * DH + t * 16, C_IN, k0, lane);
      acc[t] = wmma_bf16(a, b, acc[t]);
    }
  }

  const int nr = lane & 15;
  const int rb = (lane & 16) ? 8 : 0;
  if (which < 2) {
    bf16_t* dst = (which == 0) ? qh : kh;
#pragma unroll
    for (int t = 0; t < 2; ++t) {
      const int dm = t * 16 + nr;                     // 0..31
      // inv_freq = 10000^(-dm/32) = exp(-dm * ln(10000)/32)
      const float inv = __expf(-(float)dm * 0.28782313662425572f);
#pragma unroll
      for (int r = 0; r < 8; ++r) {
        const int m = m0 + rb + r;
        const int s = m & (SEQ - 1);
        const int b = m >> 11;
        float sn, cs;
        __sincosf((float)s * inv, &sn, &cs);
        const float lo = acc[t][r], hi = acc[t + 2][r];
        const float nlo = lo * cs - hi * sn;
        const float nhi = hi * cs + lo * sn;
        const size_t row = ((size_t)(b * NH + h) * SEQ + s) * DH;
        dst[row + dm]      = (bf16_t)nlo;
        dst[row + dm + 32] = (bf16_t)nhi;
      }
    }
  } else {
    // V stored transposed: vhT[b,h,d,s] so attn@V B-frags are contiguous.
#pragma unroll
    for (int t = 0; t < 4; ++t) {
      const int d = t * 16 + nr;
#pragma unroll
      for (int r = 0; r < 8; ++r) {
        const int m = m0 + rb + r;
        const int s = m & (SEQ - 1);
        const int b = m >> 11;
        vhT[((size_t)(b * NH + h) * DH + d) * SEQ + s] = (bf16_t)acc[t][r];
      }
    }
  }
}

// ---- kernel 2: fused scores + softmax + attn@V ----------------------------
// grid = (SEQ/16 = 128 row tiles, 1, 32 bh); block = 256 (8 waves).
// Dynamic LDS: sc[16][SC_LD] f32 score strip (~128 KB).
__global__ __launch_bounds__(256) void fused_attn_kernel(
    const bf16_t* __restrict__ qh, const bf16_t* __restrict__ kh,
    const bf16_t* __restrict__ vhT, float* __restrict__ attn,
    bf16_t* __restrict__ ctx) {
  extern __shared__ float smem[];
  float* sc = smem;                   // [16][SC_LD] scores -> softmax rows

  const int lane = threadIdx.x & 31;
  const int wave = threadIdx.x >> 5;
  const int bh   = blockIdx.z;
  const int i0   = blockIdx.x * 16;                   // row tile in this head
  const bf16_t* qb = qh  + (size_t)bh * SEQ * DH;
  const bf16_t* kb = kh  + (size_t)bh * SEQ * DH;
  const bf16_t* Vt = vhT + (size_t)bh * DH * SEQ;
  const int nr = lane & 15;
  const int rb = (lane & 16) ? 8 : 0;

  // ---- phase 1: scores into LDS.  Wave w owns cols [w*256, w*256+256). ----
  v16bf a0 = load_a_bf16(qb, i0, DH, 0, lane);        // q rows, K=0..31
  v16bf a1 = load_a_bf16(qb, i0, DH, 32, lane);       // q rows, K=32..63
#pragma unroll
  for (int jc = 0; jc < 4; ++jc) {
    const int j0 = wave * 256 + jc * 64;
    v8f acc[4] = {};
#pragma unroll
    for (int t = 0; t < 4; ++t) {
      v16bf b0 = load_b_bf16(kb, j0 + t * 16, DH, 0, lane);
      acc[t] = wmma_bf16(a0, b0, acc[t]);
      v16bf b1 = load_b_bf16(kb, j0 + t * 16, DH, 32, lane);
      acc[t] = wmma_bf16(a1, b1, acc[t]);
    }
#pragma unroll
    for (int t = 0; t < 4; ++t)
#pragma unroll
      for (int r = 0; r < 8; ++r)
        sc[(rb + r) * SC_LD + j0 + t * 16 + nr] = acc[t][r] * 0.125f;
  }
  __syncthreads();

  // ---- phase 2: softmax over LDS rows (wave handles 2 rows), then hand the
  // normalized rows to the async engine (LDS -> HBM) so the 537 MB attn store
  // overlaps the phase-3 WMMA loop. -----------------------------------------
  float* attn_out = attn + (size_t)bh * SEQ * SEQ;
#pragma unroll
  for (int rr = 0; rr < 2; ++rr) {
    const int row = wave * 2 + rr;
    float* rp = sc + row * SC_LD;
    float m = -__builtin_inff();
    for (int c = 0; c < 64; ++c) m = fmaxf(m, rp[lane + 32 * c]);
#pragma unroll
    for (int d = 16; d >= 1; d >>= 1) m = fmaxf(m, __shfl_xor(m, d, 32));
    float s = 0.f;
    for (int c = 0; c < 64; ++c) {
      const float e = __expf(rp[lane + 32 * c] - m);
      rp[lane + 32 * c] = e;
      s += e;
    }
#pragma unroll
    for (int d = 16; d >= 1; d >>= 1) s += __shfl_xor(s, d, 32);
    const float inv = 1.0f / s;
    for (int c = 0; c < 64; ++c) rp[lane + 32 * c] *= inv;   // normalize in LDS

    // drain this wave's DS writes, then async-copy the row to global attn.
    asm volatile("s_wait_dscnt 0x0" ::: "memory");
    const unsigned rp_off = (unsigned)(size_t)rp;   // LDS addr = addr[31:0]
    float* gout = attn_out + (size_t)(i0 + row) * SEQ;
    for (int c2 = 0; c2 < 16; ++c2) {
      const int e0 = c2 * 128 + lane * 4;           // 16 B per lane
      async_store_b128(gout + e0, rp_off + (unsigned)e0 * 4u);
    }
  }
  __syncthreads();

  // ---- phase 3: ctx = softmax @ V.  Wave w owns K range [w*256, w*256+256).
  v8f acc[4] = {};
  const int kbase = wave * 256;
  for (int kk = 0; kk < 256; kk += 32) {
    const int k0 = kbase + kk;
    __builtin_prefetch(Vt + (size_t)nr * SEQ + k0 + 256, 0, 0);
    v16bf a = lds_load_a_f32(sc, SC_LD, k0, lane);  // softmax rows from LDS
#pragma unroll
    for (int t = 0; t < 4; ++t) {
      v16bf b = load_b_bf16(Vt, t * 16, SEQ, k0, lane);
      acc[t] = wmma_bf16(a, b, acc[t]);
    }
  }
  // async stores must finish before sc is overwritten by the reduction.
  asm volatile("s_wait_asynccnt 0x0" ::: "memory");
  __syncthreads();
  // cross-wave reduction of 16x64 partials through LDS (reusing sc space)
#pragma unroll
  for (int t = 0; t < 4; ++t)
#pragma unroll
    for (int r = 0; r < 8; ++r)
      sc[wave * 1024 + (rb + r) * 64 + t * 16 + nr] = acc[t][r];
  __syncthreads();

  const int b = bh >> 4, h = bh & 15;
  for (int o = threadIdx.x; o < 1024; o += 256) {
    const int row = o >> 6, d = o & 63;
    float sum = 0.f;
#pragma unroll
    for (int w = 0; w < 8; ++w) sum += sc[w * 1024 + o];
    ctx[(size_t)(b * SEQ + i0 + row) * C_IN + h * DH + d] = (bf16_t)sum;
  }
}

// ---- kernel 3: out = ctx @ Wo^T -------------------------------------------
// grid = (32, 16); block = 256.  Wave: 16(m) x 64(n), K = 1024.
__global__ __launch_bounds__(256) void oproj_kernel(
    const bf16_t* __restrict__ ctx, const float* __restrict__ Wo,
    float* __restrict__ out) {
  const int lane = threadIdx.x & 31;
  const int wave = threadIdx.x >> 5;
  const int m0   = (blockIdx.x * 8 + wave) * 16;
  const int n0   = blockIdx.y * 64;

  v8f acc[4] = {};
  for (int k0 = 0; k0 < C_IN; k0 += 32) {
    v16bf a = load_a_bf16(ctx, m0, C_IN, k0, lane);
#pragma unroll
    for (int t = 0; t < 4; ++t) {
      v16bf b = load_b_f32(Wo, n0 + t * 16, C_IN, k0, lane);
      acc[t] = wmma_bf16(a, b, acc[t]);
    }
  }
  const int rb = (lane & 16) ? 8 : 0;
#pragma unroll
  for (int t = 0; t < 4; ++t)
#pragma unroll
    for (int r = 0; r < 8; ++r)
      out[(size_t)(m0 + rb + r) * C_IN + n0 + t * 16 + (lane & 15)] = acc[t][r];
}

// ---- host launch ----------------------------------------------------------
extern "C" void kernel_launch(void* const* d_in, const int* in_sizes, int n_in,
                              void* d_out, int out_size, void* d_ws,
                              size_t ws_size, hipStream_t stream) {
  const float* q  = (const float*)d_in[0];
  const float* k  = (const float*)d_in[1];
  const float* v  = (const float*)d_in[2];
  const float* Wq = (const float*)d_in[3];
  const float* Wk = (const float*)d_in[4];
  const float* Wv = (const float*)d_in[5];
  const float* Wo = (const float*)d_in[6];

  float* out  = (float*)d_out;
  float* attn = out + (size_t)2 * SEQ * C_IN;  // attn region of d_out

  // workspace: 4 x (2*16*2048*64) bf16 = 32 MB
  bf16_t* qh  = (bf16_t*)d_ws;
  bf16_t* kh  = qh + (size_t)4 * 1024 * 1024;
  bf16_t* vhT = kh + (size_t)4 * 1024 * 1024;
  bf16_t* ctx = vhT + (size_t)4 * 1024 * 1024;

  const size_t lds_bytes = (size_t)(16 * SC_LD) * sizeof(float);  // ~128 KB

  qkv_rope_kernel<<<dim3(32, NH, 3), 256, 0, stream>>>(q, k, v, Wq, Wk, Wv, qh,
                                                       kh, vhT);
  fused_attn_kernel<<<dim3(SEQ / 16, 1, 2 * NH), 256, lds_bytes, stream>>>(
      qh, kh, vhT, attn, ctx);
  oproj_kernel<<<dim3(32, NH), 256, 0, stream>>>(ctx, Wo, out);
}
